// NTXentLossClean_85624468013600
// MI455X (gfx1250) — matrix operand order repaired and checked by reading, not compiled, not run on hardware
//
#include <hip/hip_runtime.h>
#include <hip/hip_bf16.h>

#define B_N 8192
#define D_K 128
#define N_TILES (B_N / 16)

typedef __attribute__((ext_vector_type(16))) _Float16 v16h;
typedef __attribute__((ext_vector_type(8)))  _Float16 v8h;
typedef __attribute__((ext_vector_type(4)))  _Float16 v4h;
typedef __attribute__((ext_vector_type(8)))  float    v8f;
typedef __attribute__((ext_vector_type(4)))  int      v4i;

// ---- CDNA5 async global->LDS copy (guarded; falls back to sync staging) ----
#if defined(__has_builtin)
#if __has_builtin(__builtin_amdgcn_global_load_async_to_lds_b128)
#define HAVE_ASYNC_COPY 1
#endif
#if __has_builtin(__builtin_amdgcn_s_wait_asynccnt)
#define HAVE_WAIT_ASYNC 1
#endif
#endif

__device__ __forceinline__ void stage_copy16(const _Float16* g, _Float16* l)
{
#if HAVE_ASYNC_COPY
    // Builtin expects 128-bit (int4) payload pointers: global side AS(1),
    // LDS side AS(3). C-style casts drop const and perform the addrspace cast;
    // specific->generic narrows implicitly if the declared param is generic.
    __builtin_amdgcn_global_load_async_to_lds_b128(
        (__attribute__((address_space(1))) v4i*)(v4i*)g,
        (__attribute__((address_space(3))) v4i*)(v4i*)l,
        0, 0);
#else
    *reinterpret_cast<v8h*>(l) = *reinterpret_cast<const v8h*>(g);
#endif
}

__device__ __forceinline__ void wait_async_le2()
{
#if HAVE_ASYNC_COPY
#if HAVE_WAIT_ASYNC
    __builtin_amdgcn_s_wait_asynccnt(2);
#else
    asm volatile("s_wait_asynccnt 0x2" ::: "memory");
#endif
#endif
}

__device__ __forceinline__ void wait_async_le0()
{
#if HAVE_ASYNC_COPY
#if HAVE_WAIT_ASYNC
    __builtin_amdgcn_s_wait_asynccnt(0);
#else
    asm volatile("s_wait_asynccnt 0x0" ::: "memory");
#endif
#endif
}

// ---------------------------------------------------------------------------
// Phase 1: row-wise L2 normalize (torch-style: x / max(||x||, eps)) into f16,
// and pos[i] = 2 * <zi_n[i], zj_n[i]> in fp32. One wave32 per row.
// ---------------------------------------------------------------------------
__global__ __launch_bounds__(256) void normalize_pos_kernel(
    const float* __restrict__ zis, const float* __restrict__ zjs,
    _Float16* __restrict__ zi_h, _Float16* __restrict__ zj_h,
    float* __restrict__ pos)
{
    const int row  = (blockIdx.x * blockDim.x + threadIdx.x) >> 5;
    const int lane = threadIdx.x & 31;

    const float4 a = reinterpret_cast<const float4*>(zis + (size_t)row * D_K)[lane];
    const float4 b = reinterpret_cast<const float4*>(zjs + (size_t)row * D_K)[lane];

    float sa = a.x * a.x + a.y * a.y + a.z * a.z + a.w * a.w;
    float sb = b.x * b.x + b.y * b.y + b.z * b.z + b.w * b.w;
#pragma unroll
    for (int m = 16; m >= 1; m >>= 1) {
        sa += __shfl_xor(sa, m, 32);
        sb += __shfl_xor(sb, m, 32);
    }
    const float ra = 1.0f / fmaxf(sqrtf(sa), 1e-8f);
    const float rb = 1.0f / fmaxf(sqrtf(sb), 1e-8f);

    const float anx = a.x * ra, any = a.y * ra, anz = a.z * ra, anw = a.w * ra;
    const float bnx = b.x * rb, bny = b.y * rb, bnz = b.z * rb, bnw = b.w * rb;

    float d = anx * bnx + any * bny + anz * bnz + anw * bnw;
#pragma unroll
    for (int m = 16; m >= 1; m >>= 1) d += __shfl_xor(d, m, 32);

    v4h ah; ah[0] = (_Float16)anx; ah[1] = (_Float16)any; ah[2] = (_Float16)anz; ah[3] = (_Float16)anw;
    v4h bh; bh[0] = (_Float16)bnx; bh[1] = (_Float16)bny; bh[2] = (_Float16)bnz; bh[3] = (_Float16)bnw;
    reinterpret_cast<v4h*>(zi_h + (size_t)row * D_K)[lane] = ah;
    reinterpret_cast<v4h*>(zj_h + (size_t)row * D_K)[lane] = bh;

    if (lane == 0) pos[row] = 2.0f * d;   // /TEMP with TEMP=0.5
}

// ---------------------------------------------------------------------------
// Fragment loaders for CDNA5 V_WMMA_F32_16X16X32_F16, contiguous k order (same in A and B).
// A (16x32, MxK): lane<16 -> M=lane, K=kb+0..kb+15 contiguous
//                 lane>=16 -> M=lane-16, K=kb+16..kb+31 contiguous
// B (32x16, KxN): lane<16 -> N=lane, K=kb+0..kb+15 contiguous
//                 lane>=16 -> N=lane-16, K=kb+16..kb+31 contiguous
// ---------------------------------------------------------------------------
__device__ __forceinline__ v16h load_a_frag(const _Float16* __restrict__ rowp,
                                            int kbase, int koff /*0 or 16*/)
{
    const v8h lo = *reinterpret_cast<const v8h*>(rowp + kbase + koff);
    const v8h hi = *reinterpret_cast<const v8h*>(rowp + kbase + 8 + koff);
    v16h r;
#pragma unroll
    for (int i = 0; i < 8; ++i) { r[i] = lo[i]; r[i + 8] = hi[i]; }
    return r;
}

__device__ __forceinline__ v16h load_b_frag_lds(const _Float16* p /*16 contiguous*/)
{
    const v8h lo = *reinterpret_cast<const v8h*>(p);
    const v8h hi = *reinterpret_cast<const v8h*>(p + 8);
    v16h r;
#pragma unroll
    for (int i = 0; i < 8; ++i) { r[i] = lo[i]; r[i + 8] = hi[i]; }
    return r;
}

// ---------------------------------------------------------------------------
// Phase 2: fused dual-GEMM + masked exp-sum + logsumexp per row.
// 8 waves/WG; each wave owns a 16-row tile of A (in registers, K=128).
// Column tiles (16 rows x 128 f16 of zj_h and zi_h = 2 x 4KB, contiguous) are
// double-buffered through LDS with async global->LDS b128 copies: stage tile
// t+1 while all waves run 8 WMMAs + exp epilogue on tile t from LDS.
// All logits <= 2.0, so log(sum(exp)) needs no max-shift.
// ---------------------------------------------------------------------------
__global__ __launch_bounds__(256) void ntxent_wmma_kernel(
    const _Float16* __restrict__ zi_h, const _Float16* __restrict__ zj_h,
    const int* __restrict__ ids, const float* __restrict__ pos,
    float* __restrict__ row_terms)
{
    __shared__ _Float16 sj[2][16 * D_K];   // zj column tile, double buffered
    __shared__ _Float16 si[2][16 * D_K];   // zi column tile, double buffered

    const int tid     = threadIdx.x;
    const int lane    = tid & 31;
    const int wave    = tid >> 5;
    const int rowBase = (blockIdx.x * 8 + wave) * 16;
    const int nIdx    = lane & 15;
    const bool hiHalf = lane >= 16;
    const int off     = hiHalf ? 8 : 0;    // C/D: M = r + off for VGPR r
    const int aKoff   = hiHalf ? 16 : 0;
    const int bKhalf  = hiHalf ? 16 : 0;

    // A fragments: row (rowBase + nIdx) of normalized zis, full K=128.
    const _Float16* arow = zi_h + (size_t)(rowBase + nIdx) * D_K;
    v16h afrag[4];
#pragma unroll
    for (int kc = 0; kc < 4; ++kc) afrag[kc] = load_a_frag(arow, kc * 32, aKoff);

    int idRow[8];
#pragma unroll
    for (int r = 0; r < 8; ++r) idRow[r] = ids[rowBase + r + off];

    float acc[8];
#pragma unroll
    for (int r = 0; r < 8; ++r) acc[r] = 0.0f;

    // Stage tile 0 (each thread copies 16B of zj tile + 16B of zi tile).
    {
        const size_t gOff = (size_t)tid * 8;
        stage_copy16(zj_h + gOff, &sj[0][0] + gOff);
        stage_copy16(zi_h + gOff, &si[0][0] + gOff);
    }

    for (int t = 0; t < N_TILES; ++t) {
        const int cur = t & 1;
        const int colBase = t * 16;

        if (t + 1 < N_TILES) {
            const int nxt = cur ^ 1;
            const size_t gBase = (size_t)(colBase + 16) * D_K + (size_t)tid * 8;
            stage_copy16(zj_h + gBase, &sj[nxt][0] + (size_t)tid * 8);
            stage_copy16(zi_h + gBase, &si[nxt][0] + (size_t)tid * 8);
            wait_async_le2();     // current tile's 2 copies retired (in-order)
        } else {
            wait_async_le0();
        }
        __syncthreads();          // current tile visible to all waves

        const _Float16* bjRow = &sj[cur][0] + nIdx * D_K + bKhalf;
        const _Float16* biRow = &si[cur][0] + nIdx * D_K + bKhalf;

        v8f cj = {};  // zi_n x zj_n^T tile
        v8f ck = {};  // zi_n x zi_n^T tile
#pragma unroll
        for (int kc = 0; kc < 4; ++kc) {
            const v16h bj = load_b_frag_lds(bjRow + kc * 32);
            cj = __builtin_amdgcn_wmma_f32_16x16x32_f16(
                false, afrag[kc], false, bj, (short)0, cj, false, false);
            const v16h bi = load_b_frag_lds(biRow + kc * 32);
            ck = __builtin_amdgcn_wmma_f32_16x16x32_f16(
                false, afrag[kc], false, bi, (short)0, ck, false, false);
        }

        const int col = colBase + nIdx;
        const int idc = ids[col];
#pragma unroll
        for (int r = 0; r < 8; ++r) {
            const int row = rowBase + r + off;
            // negatives: off-diagonal AND different partnet id (branchless:
            // keeps EXEC fully set around the WMMA loop)
            const float m = ((col != row) && (idc != idRow[r])) ? 1.0f : 0.0f;
            acc[r] += m * (__expf(2.0f * cj[r]) + __expf(2.0f * ck[r]));
        }

        __syncthreads();          // all waves done with cur before restaging it
    }

    // Sum over the 16 N-lanes of each half (xor masks 1..8 stay within half).
#pragma unroll
    for (int r = 0; r < 8; ++r) {
        float v = acc[r];
        v += __shfl_xor(v, 1, 32);
        v += __shfl_xor(v, 2, 32);
        v += __shfl_xor(v, 4, 32);
        v += __shfl_xor(v, 8, 32);
        acc[r] = v;
    }

    if (nIdx == 0) {
#pragma unroll
        for (int r = 0; r < 8; ++r) {
            const int row = rowBase + r + off;
            const float p = pos[row];
            const float total = acc[r] + __expf(p);   // positive logit always kept
            row_terms[row] = __logf(total) - p;       // lse - pos
        }
    }
}

// ---------------------------------------------------------------------------
// Phase 3: deterministic single-block reduction of the 8192 per-row terms.
// ---------------------------------------------------------------------------
__global__ __launch_bounds__(256) void final_reduce_kernel(
    const float* __restrict__ row_terms, float* __restrict__ out)
{
    __shared__ float sm[256];
    float s = 0.0f;
    for (int i = threadIdx.x; i < B_N; i += 256) s += row_terms[i];
    sm[threadIdx.x] = s;
    __syncthreads();
#pragma unroll
    for (int stride = 128; stride > 0; stride >>= 1) {
        if (threadIdx.x < stride) sm[threadIdx.x] += sm[threadIdx.x + stride];
        __syncthreads();
    }
    if (threadIdx.x == 0) out[0] = sm[0] / (2.0f * (float)B_N);
}

extern "C" void kernel_launch(void* const* d_in, const int* in_sizes, int n_in,
                              void* d_out, int out_size, void* d_ws, size_t ws_size,
                              hipStream_t stream)
{
    (void)in_sizes; (void)n_in; (void)out_size; (void)ws_size;
    const float* zis = (const float*)d_in[0];
    const float* zjs = (const float*)d_in[1];
    const int*   ids = (const int*)d_in[2];
    float* out = (float*)d_out;

    char* ws = (char*)d_ws;
    _Float16* zi_h      = (_Float16*)(ws);
    _Float16* zj_h      = (_Float16*)(ws + (size_t)B_N * D_K * 2);
    float*    pos       = (float*)   (ws + (size_t)B_N * D_K * 4);
    float*    row_terms = (float*)   (ws + (size_t)B_N * D_K * 4 + (size_t)B_N * 4);

    normalize_pos_kernel<<<B_N / 8, 256, 0, stream>>>(zis, zjs, zi_h, zj_h, pos);
    ntxent_wmma_kernel<<<B_N / 128, 256, 0, stream>>>(zi_h, zj_h, ids, pos, row_terms);
    final_reduce_kernel<<<1, 256, 0, stream>>>(row_terms, out);
}
